// GraphODEFunc_9740985827791
// MI455X (gfx1250) — compile-verified, matrix operand order checked by reading, not run on hardware
//
#include <hip/hip_runtime.h>
#include <math.h>

#define BB 16
#define NN 1024
#define HH 128
#define MASK_EPS 1e-9f

typedef __attribute__((ext_vector_type(16))) __bf16 v16bf;
typedef __attribute__((ext_vector_type(8)))  __bf16 v8bf;
typedef __attribute__((ext_vector_type(8)))  float  v8f;
typedef __attribute__((ext_vector_type(4)))  float  v4f;

__device__ __forceinline__ v8f wmma_bf16(v16bf a, v16bf b, v8f c) {
    return __builtin_amdgcn_wmma_f32_16x16x32_bf16(false, a, false, b, (short)0, c, false, false);
}

// Load 16 bf16 operand elements for one lane: elements j<8 at base[off0+j],
// j>=8 at base[off0+16+j-8]  (K-mapping for 16-bit WMMA operands, off0 = k0+8*hi)
__device__ __forceinline__ v16bf load_bf16_op(const __bf16* base, int off0) {
    v8bf lo = *(const v8bf*)(base + off0);
    v8bf hi = *(const v8bf*)(base + off0 + 16);
    return __builtin_shufflevector(lo, hi, 0, 1, 2, 3, 4, 5, 6, 7,
                                   8, 9, 10, 11, 12, 13, 14, 15);
}

__device__ __forceinline__ v16bf load_f32_op(const float* base, int off0) {
    v4f f0 = *(const v4f*)(base + off0);
    v4f f1 = *(const v4f*)(base + off0 + 4);
    v4f f2 = *(const v4f*)(base + off0 + 16);
    v4f f3 = *(const v4f*)(base + off0 + 20);
    v16bf r;
#pragma unroll
    for (int j = 0; j < 4; ++j) {
        r[j]      = (__bf16)f0[j];
        r[j + 4]  = (__bf16)f1[j];
        r[j + 8]  = (__bf16)f2[j];
        r[j + 12] = (__bf16)f3[j];
    }
    return r;
}

// ---------------------------------------------------------------------------
// GEMM: HT[b, o, n] = sum_d X[b*N+n, d] * W[o, d]
// X: [B*N, H] fp32 or bf16 (row-major, K contiguous); output feature-major.
// One wave per 16x16 output tile; bf16 WMMA, fp32 accum.
// ---------------------------------------------------------------------------
template <bool IN_F32>
__global__ __launch_bounds__(256) void k_gemm_xw(const void* __restrict__ Xv,
                                                 const float* __restrict__ W,
                                                 __bf16* __restrict__ HT) {
    int wave = (blockIdx.x * blockDim.x + threadIdx.x) >> 5;
    int lane = threadIdx.x & 31;
    int tc = wave % (HH / 16);
    int tr = wave / (HH / 16);           // 0 .. B*N/16-1
    int row0 = tr * 16;                  // flattened b*N + n, tile within one b
    int c0 = tc * 16;
    int m = lane & 15, hi = lane >> 4;
    int off0 = 8 * hi;
    const float* Xf = (const float*)Xv + (size_t)(row0 + m) * HH;
    const __bf16* Xb = (const __bf16*)Xv + (size_t)(row0 + m) * HH;
    const float* Wr = W + (size_t)(c0 + m) * HH;
    v8f acc = {};
    for (int k0 = 0; k0 < HH; k0 += 32) {
        v16bf a = IN_F32 ? load_f32_op(Xf, k0 + off0)
                         : load_bf16_op(Xb, k0 + off0);
        v16bf b = load_f32_op(Wr, k0 + off0);
        acc = wmma_bf16(a, b, acc);
    }
    // C element: M = v + 8*hi (row), N = m (col).  Transposed store:
    // HT[(b*H + c0+m)*N + n] with n = nbase + v + 8*hi  -> 8 consecutive -> b128
    int b = row0 / NN;
    int nbase = row0 % NN;
    v8bf st;
#pragma unroll
    for (int v = 0; v < 8; ++v) st[v] = (__bf16)acc[v];
    *(v8bf*)(&HT[((size_t)b * HH + c0 + m) * NN + nbase + 8 * hi]) = st;
}

// ---------------------------------------------------------------------------
// attention scores from feature-major hT: S[b,n] = sum_h hT[b,h,n]*a_src[h]
// ---------------------------------------------------------------------------
__global__ __launch_bounds__(256) void k_scores(const __bf16* __restrict__ HT,
                                                const float* __restrict__ a_src,
                                                const float* __restrict__ a_dst,
                                                float* __restrict__ S,
                                                float* __restrict__ D) {
    int wave = (blockIdx.x * blockDim.x + threadIdx.x) >> 5;  // b*N+n
    int lane = threadIdx.x & 31;
    int b = wave / NN, n = wave % NN;
    float s = 0.f, d = 0.f;
    for (int h = lane; h < HH; h += 32) {
        float hv = (float)HT[((size_t)b * HH + h) * NN + n];
        s += hv * a_src[h];
        d += hv * a_dst[h];
    }
#pragma unroll
    for (int off = 16; off; off >>= 1) {
        s += __shfl_down(s, off);
        d += __shfl_down(d, off);
    }
    if (lane == 0) { S[wave] = s; D[wave] = d; }
}

// ---------------------------------------------------------------------------
// Fused masked softmax + WMMA aggregation for one GAT.
// grid (N/16, B), block 256 (8 waves). alpha staged as bf16 in LDS.
// HT is feature-major [B,H,N] so the B-operand is K-contiguous (b128 loads).
// ---------------------------------------------------------------------------
__global__ __launch_bounds__(256) void k_gat_attn(const float* __restrict__ A,
                                                  const float* __restrict__ Srow,
                                                  const float* __restrict__ Dcol,
                                                  const __bf16* __restrict__ HT,
                                                  float* __restrict__ OutF,
                                                  __bf16* __restrict__ OutBf) {
    __shared__ __bf16 sAlpha[16][NN];
    __shared__ float sMax[16], sSum[16];
    int b = blockIdx.y;
    int n0 = blockIdx.x * 16;
    int tid = threadIdx.x, lane = tid & 31, wid = tid >> 5;

    // pass 1: streaming softmax stats; wave handles 2 rows
    for (int rr = 0; rr < 2; ++rr) {
        int r = wid * 2 + rr;
        float s = Srow[b * NN + n0 + r];
        float mx = -INFINITY;
        for (int mm = lane; mm < NN; mm += 32) {
            float aij = A[(size_t)(n0 + r) * NN + mm];
            if (aij >= MASK_EPS) {
                float e = s + Dcol[b * NN + mm];
                e = (e > 0.f) ? e : 0.2f * e;
                e *= 0.5f; // / TAU
                mx = fmaxf(mx, e);
            }
        }
#pragma unroll
        for (int off = 16; off; off >>= 1) mx = fmaxf(mx, __shfl_xor(mx, off));
        float sum = 0.f;
        for (int mm = lane; mm < NN; mm += 32) {
            float aij = A[(size_t)(n0 + r) * NN + mm];
            if (aij >= MASK_EPS) {
                float e = s + Dcol[b * NN + mm];
                e = (e > 0.f) ? e : 0.2f * e;
                e *= 0.5f;
                sum += __expf(e - mx);
            }
        }
#pragma unroll
        for (int off = 16; off; off >>= 1) sum += __shfl_xor(sum, off);
        if (lane == 0) { sMax[r] = mx; sSum[r] = sum; }
    }
    __syncthreads();

    // pass 2: materialize alpha (bf16) cooperatively into LDS
    for (int idx = tid; idx < 16 * NN; idx += 256) {
        int r = idx >> 10, mm = idx & (NN - 1);
        float aij = A[(size_t)(n0 + r) * NN + mm];
        float al = 0.f;
        float sum = sSum[r];
        if (aij >= MASK_EPS && sum > 0.f) {
            float e = Srow[b * NN + n0 + r] + Dcol[b * NN + mm];
            e = (e > 0.f) ? e : 0.2f * e;
            e *= 0.5f;
            al = __expf(e - sMax[r]) / sum;
        }
        sAlpha[r][mm] = (__bf16)al;
    }
    __syncthreads();

    // pass 3: C[16x16] = alpha[16xN] * h[Nx16-slice] via WMMA; wave wid -> cols
    int c0 = wid * 16;
    int m = lane & 15, hi = lane >> 4;
    int off0 = 8 * hi;
    const __bf16* bptr = HT + ((size_t)b * HH + c0 + m) * NN;
    const __bf16* aptr = &sAlpha[m][0];
    v8f acc = {};
    for (int k0 = 0; k0 < NN; k0 += 32) {
        if (k0 + 32 < NN) __builtin_prefetch(bptr + k0 + 32, 0, 1);
        v16bf a = load_bf16_op(aptr, k0 + off0);
        v16bf bm = load_bf16_op(bptr, k0 + off0);
        acc = wmma_bf16(a, bm, acc);
    }
#pragma unroll
    for (int v = 0; v < 8; ++v) {
        int M = v + 8 * hi;
        float val = tanhf(acc[v]);
        size_t o = ((size_t)b * NN + n0 + M) * HH + c0 + m;
        if (OutF) OutF[o] = val;
        if (OutBf) OutBf[o] = (__bf16)val;
    }
}

// ---------------------------------------------------------------------------
// hx = H_conv @ (x@theta): A = H_conv (f32, K contiguous), B = xtT (bf16,
// feature-major, K contiguous). Output row-major fp32.
// ---------------------------------------------------------------------------
__global__ __launch_bounds__(256) void k_hconv(const float* __restrict__ Hc,
                                               const __bf16* __restrict__ XtT,
                                               float* __restrict__ Hx) {
    int wave = (blockIdx.x * blockDim.x + threadIdx.x) >> 5;
    int lane = threadIdx.x & 31;
    int tc = wave % (HH / 16);
    int t2 = wave / (HH / 16);
    int trow = t2 % (NN / 16);
    int b = t2 / (NN / 16);
    int n0 = trow * 16, c0 = tc * 16;
    int m = lane & 15, hi = lane >> 4;
    int off0 = 8 * hi;
    const float* aptr = Hc + (size_t)(n0 + m) * NN;
    const __bf16* bptr = XtT + ((size_t)b * HH + c0 + m) * NN;
    v8f acc = {};
    for (int k0 = 0; k0 < NN; k0 += 32) {
        if (k0 + 32 < NN) __builtin_prefetch(bptr + k0 + 32, 0, 1);
        v16bf a = load_f32_op(aptr, k0 + off0);
        v16bf bm = load_bf16_op(bptr, k0 + off0);
        acc = wmma_bf16(a, bm, acc);
    }
#pragma unroll
    for (int v = 0; v < 8; ++v) {
        int M = v + 8 * hi;
        Hx[((size_t)b * NN + n0 + M) * HH + c0 + m] = acc[v];
    }
}

// ---------------------------------------------------------------------------
// Final fused mix-MLP + gating + LayerNorm. One block (128 threads) per (b,n).
// ---------------------------------------------------------------------------
__global__ __launch_bounds__(128) void k_final(const float* __restrict__ h_sym,
                                               const float* __restrict__ h_fwd,
                                               const float* __restrict__ h_bwd,
                                               const float* __restrict__ h_corr,
                                               const float* __restrict__ hx,
                                               const float* __restrict__ mixW1,
                                               const float* __restrict__ mixb1,
                                               const float* __restrict__ mixW2,
                                               const float* __restrict__ mixb2,
                                               const float* __restrict__ gate,
                                               const float* __restrict__ ln_g,
                                               const float* __restrict__ ln_b,
                                               float* __restrict__ out) {
    __shared__ float c[4 * HH];
    __shared__ float r4x[128], r4y[128], r4z[128], r4w[128];
    __shared__ float red[128];
    int bn = blockIdx.x;
    int n = bn % NN;
    int h = threadIdx.x;
    size_t base = (size_t)bn * HH;
    float vs = h_sym[base + h], vf = h_fwd[base + h];
    float vb = h_bwd[base + h], vc = h_corr[base + h];
    c[h] = vs; c[HH + h] = vf; c[2 * HH + h] = vb; c[3 * HH + h] = vc;
    __syncthreads();
    float acc = mixb1[h];
    for (int f = 0; f < 4 * HH; ++f) acc += mixW1[(size_t)h * 4 * HH + f] * c[f];
    float zh = tanhf(acc);
    r4x[h] = mixW2[0 * HH + h] * zh;
    r4y[h] = mixW2[1 * HH + h] * zh;
    r4z[h] = mixW2[2 * HH + h] * zh;
    r4w[h] = mixW2[3 * HH + h] * zh;
    __syncthreads();
    for (int off = 64; off; off >>= 1) {
        if (h < off) {
            r4x[h] += r4x[h + off]; r4y[h] += r4y[h + off];
            r4z[h] += r4z[h + off]; r4w[h] += r4w[h + off];
        }
        __syncthreads();
    }
    float l0 = r4x[0] + mixb2[0], l1 = r4y[0] + mixb2[1];
    float l2 = r4z[0] + mixb2[2], l3 = r4w[0] + mixb2[3];
    float lm = fmaxf(fmaxf(l0, l1), fmaxf(l2, l3));
    float e0 = __expf(l0 - lm), e1 = __expf(l1 - lm);
    float e2 = __expf(l2 - lm), e3 = __expf(l3 - lm);
    float inv = 1.f / (e0 + e1 + e2 + e3);
    float g = 1.f / (1.f + __expf(-gate[n]));
    float hv = (e0 * vs + e1 * vf + e2 * vb + e3 * vc) * inv + g * tanhf(hx[base + h]);
    __syncthreads();
    red[h] = hv;
    __syncthreads();
    for (int off = 64; off; off >>= 1) {
        if (h < off) red[h] += red[h + off];
        __syncthreads();
    }
    float mu = red[0] / (float)HH;
    __syncthreads();
    red[h] = (hv - mu) * (hv - mu);
    __syncthreads();
    for (int off = 64; off; off >>= 1) {
        if (h < off) red[h] += red[h + off];
        __syncthreads();
    }
    float var = red[0] / (float)HH;
    out[base + h] = (hv - mu) * rsqrtf(var + 1e-5f) * ln_g[h] + ln_b[h];
}

// ---------------------------------------------------------------------------
extern "C" void kernel_launch(void* const* d_in, const int* in_sizes, int n_in,
                              void* d_out, int out_size, void* d_ws, size_t ws_size,
                              hipStream_t stream) {
    (void)in_sizes; (void)n_in; (void)out_size; (void)ws_size;
    const float* x       = (const float*)d_in[1];
    const float* A_sym   = (const float*)d_in[2];
    const float* A_fwd   = (const float*)d_in[3];
    const float* A_bwd   = (const float*)d_in[4];
    const float* A_corr  = (const float*)d_in[5];
    const float* H_conv  = (const float*)d_in[6];
    const float* W_sym1  = (const float*)d_in[7];
    const float* as1     = (const float*)d_in[8];
    const float* ad1     = (const float*)d_in[9];
    const float* W_sym2  = (const float*)d_in[10];
    const float* as2     = (const float*)d_in[11];
    const float* ad2     = (const float*)d_in[12];
    const float* W_fwd   = (const float*)d_in[13];
    const float* asf     = (const float*)d_in[14];
    const float* adf     = (const float*)d_in[15];
    const float* W_bwd   = (const float*)d_in[16];
    const float* asb     = (const float*)d_in[17];
    const float* adb     = (const float*)d_in[18];
    const float* W_corr  = (const float*)d_in[19];
    const float* asc     = (const float*)d_in[20];
    const float* adc     = (const float*)d_in[21];
    const float* mixW1   = (const float*)d_in[22];
    const float* mixb1   = (const float*)d_in[23];
    const float* mixW2   = (const float*)d_in[24];
    const float* mixb2   = (const float*)d_in[25];
    const float* theta   = (const float*)d_in[26];
    const float* gate    = (const float*)d_in[27];
    const float* ln_g    = (const float*)d_in[28];
    const float* ln_b    = (const float*)d_in[29];
    float* out = (float*)d_out;

    // workspace carve-up
    const size_t BNH = (size_t)BB * NN * HH;
    char* p = (char*)d_ws;
    auto takeB = [&](size_t bytes) { char* r = p; p += (bytes + 255) & ~(size_t)255; return r; };
    __bf16* hT_sym1  = (__bf16*)takeB(BNH * 2);   // [B,H,N] feature-major
    __bf16* hT_fwd   = (__bf16*)takeB(BNH * 2);
    __bf16* hT_bwd   = (__bf16*)takeB(BNH * 2);
    __bf16* hT_corr  = (__bf16*)takeB(BNH * 2);
    __bf16* hT_theta = (__bf16*)takeB(BNH * 2);
    __bf16* hT_h2    = (__bf16*)takeB(BNH * 2);
    __bf16* h1_bf    = (__bf16*)takeB(BNH * 2);   // [B,N,H] row-major (gemm2 input)
    float* s1 = (float*)takeB(BB * NN * 4); float* d1 = (float*)takeB(BB * NN * 4);
    float* s2 = (float*)takeB(BB * NN * 4); float* d2 = (float*)takeB(BB * NN * 4);
    float* sf = (float*)takeB(BB * NN * 4); float* df = (float*)takeB(BB * NN * 4);
    float* sb = (float*)takeB(BB * NN * 4); float* db = (float*)takeB(BB * NN * 4);
    float* sc = (float*)takeB(BB * NN * 4); float* dc = (float*)takeB(BB * NN * 4);
    float* o_sym  = (float*)takeB(BNH * 4);
    float* o_fwd  = (float*)takeB(BNH * 4);
    float* o_bwd  = (float*)takeB(BNH * 4);
    float* o_corr = (float*)takeB(BNH * 4);
    float* o_hx   = (float*)takeB(BNH * 4);

    const int gemmBlocks = (BB * NN / 16) * (HH / 16) / 8;   // 1024
    const int scoreBlocks = (BB * NN) / 8;                    // 2048
    dim3 attnGrid(NN / 16, BB);

    // Stage 1: five projections x@Wt (bf16 WMMA, fp32 accum), feature-major out
    k_gemm_xw<true><<<gemmBlocks, 256, 0, stream>>>(x, W_sym1, hT_sym1);
    k_gemm_xw<true><<<gemmBlocks, 256, 0, stream>>>(x, W_fwd,  hT_fwd);
    k_gemm_xw<true><<<gemmBlocks, 256, 0, stream>>>(x, W_bwd,  hT_bwd);
    k_gemm_xw<true><<<gemmBlocks, 256, 0, stream>>>(x, W_corr, hT_corr);
    k_gemm_xw<true><<<gemmBlocks, 256, 0, stream>>>(x, theta,  hT_theta);

    // Stage 2: attention scores
    k_scores<<<scoreBlocks, 256, 0, stream>>>(hT_sym1, as1, ad1, s1, d1);
    k_scores<<<scoreBlocks, 256, 0, stream>>>(hT_fwd,  asf, adf, sf, df);
    k_scores<<<scoreBlocks, 256, 0, stream>>>(hT_bwd,  asb, adb, sb, db);
    k_scores<<<scoreBlocks, 256, 0, stream>>>(hT_corr, asc, adc, sc, dc);

    // Stage 3: masked softmax + WMMA aggregation (+ tanh)
    k_gat_attn<<<attnGrid, 256, 0, stream>>>(A_sym,  s1, d1, hT_sym1, nullptr, h1_bf);
    k_gat_attn<<<attnGrid, 256, 0, stream>>>(A_fwd,  sf, df, hT_fwd,  o_fwd,  nullptr);
    k_gat_attn<<<attnGrid, 256, 0, stream>>>(A_bwd,  sb, db, hT_bwd,  o_bwd,  nullptr);
    k_gat_attn<<<attnGrid, 256, 0, stream>>>(A_corr, sc, dc, hT_corr, o_corr, nullptr);

    // Stage 4: second sym layer
    k_gemm_xw<false><<<gemmBlocks, 256, 0, stream>>>(h1_bf, W_sym2, hT_h2);
    k_scores<<<scoreBlocks, 256, 0, stream>>>(hT_h2, as2, ad2, s2, d2);
    k_gat_attn<<<attnGrid, 256, 0, stream>>>(A_sym, s2, d2, hT_h2, o_sym, nullptr);

    // Stage 5: hypergraph conv term
    k_hconv<<<gemmBlocks, 256, 0, stream>>>(H_conv, hT_theta, o_hx);

    // Stage 6: fused mix-MLP + gate + LayerNorm
    k_final<<<BB * NN, 128, 0, stream>>>(o_sym, o_fwd, o_bwd, o_corr, o_hx,
                                         mixW1, mixb1, mixW2, mixb2, gate,
                                         ln_g, ln_b, out);
}
